// Encoder_33509334843498
// MI455X (gfx1250) — matrix operand order repaired and checked by reading, not compiled, not run on hardware
//
#include <hip/hip_runtime.h>
#include <hip/hip_bf16.h>

// ---------------------------------------------------------------------------
// CDNA5 (gfx1250) implementation of VGG11 -> vocab score -> 3x GATv2 layers.
// Heavy lifting (all convs >= conv1, FC, GAT projections) runs on
// v_wmma_f32_16x16x32_f16 with f16 activations/weights, f32 accumulate,
// and 2x2 register blocking (32x32 output tile per wave) for operand reuse.
// ---------------------------------------------------------------------------

typedef _Float16 f16x8  __attribute__((ext_vector_type(8)));
typedef _Float16 f16x16 __attribute__((ext_vector_type(16)));
typedef float    f32x8  __attribute__((ext_vector_type(8)));

#define NBATCH 1024
#define NEDGE  32768
#define NETOT  33792   /* NEDGE + NBATCH self loops */

static __device__ __forceinline__ f16x16 zero16h() {
  f16x16 z;
#pragma unroll
  for (int i = 0; i < 16; ++i) z[i] = (_Float16)0.f;
  return z;
}
static __device__ __forceinline__ f32x8 zero8f() {
  f32x8 z;
#pragma unroll
  for (int i = 0; i < 8; ++i) z[i] = 0.f;
  return z;
}

// A fragment (16x32 f16, M x K). lane group g: element e -> k = 8g + e + (e>=8 ? 8 : 0)
static __device__ __forceinline__ f16x16 load_a_frag(const _Float16* __restrict__ p, int g) {
  f16x8 lo = *(const f16x8*)(p + 8 * g);
  f16x8 hi = *(const f16x8*)(p + 8 * g + 16);
  f16x16 a;
#pragma unroll
  for (int i = 0; i < 8; ++i) { a[i] = lo[i]; a[i + 8] = hi[i]; }
  return a;
}

// B fragment (32x16 f16, K x N) from Bt stored [N][K]: element e -> k = 8g + e + (e>=8 ? 8 : 0)
static __device__ __forceinline__ f16x16 load_b_frag(const _Float16* __restrict__ p, int g) {
  f16x8 lo = *(const f16x8*)(p + 8 * g);
  f16x8 hi = *(const f16x8*)(p + 8 * g + 16);
  f16x16 b;
#pragma unroll
  for (int i = 0; i < 8; ++i) { b[i] = lo[i]; b[i + 8] = hi[i]; }
  return b;
}

#define WMMA_F16(A, B, C) \
  __builtin_amdgcn_wmma_f32_16x16x32_f16(false, (A), false, (B), (short)0, (C), false, false)

// ---------------- generic WMMA GEMM: C[M,N] = A[M,Kp] * Bt[N,Kp]^T ----------
// 2x2 register blocking: each wave produces a 32x32 tile. M%32==0, N%32==0.
__global__ void k_gemm_wmma(const _Float16* __restrict__ A, const _Float16* __restrict__ Bt,
                            float* __restrict__ C, _Float16* __restrict__ Ch,
                            int M, int N, int Kp) {
  int wave = (int)((blockIdx.x * blockDim.x + threadIdx.x) >> 5);
  int lane = threadIdx.x & 31;
  int ntiles = N >> 5;
  int tiles = (M >> 5) * ntiles;
  if (wave >= tiles) return;
  int mt = wave / ntiles, nt = wave - mt * ntiles;
  int g = lane >> 4, l = lane & 15;
  const _Float16* arow0 = A + (size_t)(mt * 32 + l) * Kp;
  const _Float16* arow1 = arow0 + (size_t)16 * Kp;
  const _Float16* brow0 = Bt + (size_t)(nt * 32 + l) * Kp;
  const _Float16* brow1 = brow0 + (size_t)16 * Kp;
  f32x8 c00 = zero8f(), c01 = zero8f(), c10 = zero8f(), c11 = zero8f();
  for (int k0 = 0; k0 < Kp; k0 += 32) {
    f16x16 a0 = load_a_frag(arow0 + k0, g);
    f16x16 a1 = load_a_frag(arow1 + k0, g);
    f16x16 b0 = load_b_frag(brow0 + k0, g);
    f16x16 b1 = load_b_frag(brow1 + k0, g);
    c00 = WMMA_F16(a0, b0, c00);
    c01 = WMMA_F16(a0, b1, c01);
    c10 = WMMA_F16(a1, b0, c10);
    c11 = WMMA_F16(a1, b1, c11);
  }
  int n0 = nt * 32 + l, n1 = n0 + 16;
#pragma unroll
  for (int v = 0; v < 8; ++v) {
    int m0 = mt * 32 + v + 8 * g, m1 = m0 + 16;
    size_t o00 = (size_t)m0 * N + n0, o01 = (size_t)m0 * N + n1;
    size_t o10 = (size_t)m1 * N + n0, o11 = (size_t)m1 * N + n1;
    C[o00] = c00[v]; C[o01] = c01[v]; C[o10] = c10[v]; C[o11] = c11[v];
    if (Ch) {
      Ch[o00] = (_Float16)c00[v]; Ch[o01] = (_Float16)c01[v];
      Ch[o10] = (_Float16)c10[v]; Ch[o11] = (_Float16)c11[v];
    }
  }
}

// ---------------- implicit-GEMM 3x3 SAME conv, NHWC f16, WMMA ---------------
// in [NB,H,W,Cin] f16, wT [Cout][9*Cin] f16 (k = (ky*3+kx)*Cin + cin),
// out [NB,H,W,Cout] f16 = relu(conv + bias). 32 pixels x 32 couts per wave.
__global__ void k_conv_wmma(const _Float16* __restrict__ in, const _Float16* __restrict__ wT,
                            const float* __restrict__ bias, _Float16* __restrict__ out,
                            int M /*NB*H*W*/, int H, int W, int Cin, int Cout) {
  int wave = (int)((blockIdx.x * blockDim.x + threadIdx.x) >> 5);
  int lane = threadIdx.x & 31;
  int ntiles = Cout >> 5;
  int tiles = (M >> 5) * ntiles;
  if (wave >= tiles) return;
  int mt = wave / ntiles, nt = wave - mt * ntiles;
  int g = lane >> 4, l = lane & 15;
  int hw = H * W;
  int p0 = mt * 32 + l, p1 = p0 + 16;
  int nb0 = p0 / hw, r0 = p0 - nb0 * hw, y0 = r0 / W, x0 = r0 - y0 * W;
  int nb1 = p1 / hw, r1 = p1 - nb1 * hw, y1 = r1 / W, x1 = r1 - y1 * W;
  int n0 = nt * 32 + l, n1 = n0 + 16;
  int K9 = 9 * Cin;
  const _Float16* brow0 = wT + (size_t)n0 * K9;
  const _Float16* brow1 = wT + (size_t)n1 * K9;
  f32x8 c00 = zero8f(), c01 = zero8f(), c10 = zero8f(), c11 = zero8f();
  for (int k0 = 0; k0 < K9; k0 += 32) {
    int tap = k0 / Cin;             // Cin multiple of 32 -> chunk stays in one tap
    int cin0 = k0 - tap * Cin;
    int ky = tap / 3, kx = tap - ky * 3;
    int iy0 = y0 + ky - 1, ix0 = x0 + kx - 1;
    int iy1 = y1 + ky - 1, ix1 = x1 + kx - 1;
    f16x16 a0, a1;
    if (iy0 >= 0 && iy0 < H && ix0 >= 0 && ix0 < W)
      a0 = load_a_frag(in + (((size_t)nb0 * H + iy0) * W + ix0) * Cin + cin0, g);
    else
      a0 = zero16h();
    if (iy1 >= 0 && iy1 < H && ix1 >= 0 && ix1 < W)
      a1 = load_a_frag(in + (((size_t)nb1 * H + iy1) * W + ix1) * Cin + cin0, g);
    else
      a1 = zero16h();
    f16x16 b0 = load_b_frag(brow0 + k0, g);
    f16x16 b1 = load_b_frag(brow1 + k0, g);
    c00 = WMMA_F16(a0, b0, c00);
    c01 = WMMA_F16(a0, b1, c01);
    c10 = WMMA_F16(a1, b0, c10);
    c11 = WMMA_F16(a1, b1, c11);
  }
  float bv0 = bias[n0], bv1 = bias[n1];
#pragma unroll
  for (int v = 0; v < 8; ++v) {
    int pm0 = mt * 32 + v + 8 * g, pm1 = pm0 + 16;
    float r00 = c00[v] + bv0; r00 = r00 > 0.f ? r00 : 0.f;
    float r01 = c01[v] + bv1; r01 = r01 > 0.f ? r01 : 0.f;
    float r10 = c10[v] + bv0; r10 = r10 > 0.f ? r10 : 0.f;
    float r11 = c11[v] + bv1; r11 = r11 > 0.f ? r11 : 0.f;
    out[(size_t)pm0 * Cout + n0] = (_Float16)r00;
    out[(size_t)pm0 * Cout + n1] = (_Float16)r01;
    out[(size_t)pm1 * Cout + n0] = (_Float16)r10;
    out[(size_t)pm1 * Cout + n1] = (_Float16)r11;
  }
}

// ---------------- conv0: 1 -> 64 channels, direct (K=9, not WMMA friendly) --
__global__ void k_conv0(const float* __restrict__ in, const float* __restrict__ w,
                        const float* __restrict__ b, _Float16* __restrict__ out) {
  int idx = blockIdx.x * blockDim.x + threadIdx.x;   // NB*32*32*64
  if (idx >= NBATCH * 32 * 32 * 64) return;
  int co = idx & 63;
  int p = idx >> 6;
  int x = p & 31, y = (p >> 5) & 31, nb = p >> 10;
  const float* ip = in + (size_t)nb * 1024;
  float acc = b[co];
#pragma unroll
  for (int ky = 0; ky < 3; ++ky) {
    int iy = y + ky - 1;
    if (iy < 0 || iy >= 32) continue;
#pragma unroll
    for (int kx = 0; kx < 3; ++kx) {
      int ix = x + kx - 1;
      if (ix < 0 || ix >= 32) continue;
      acc += ip[iy * 32 + ix] * w[co * 9 + ky * 3 + kx];
    }
  }
  out[idx] = (_Float16)(acc > 0.f ? acc : 0.f);
}

// ---------------- 2x2 max pool, NHWC f16 ------------------------------------
__global__ void k_pool(const _Float16* __restrict__ in, _Float16* __restrict__ out,
                       int Ho, int Wo, int C) {
  int idx = blockIdx.x * blockDim.x + threadIdx.x;
  int total = NBATCH * Ho * Wo * C;
  if (idx >= total) return;
  int ch = idx % C;
  int t = idx / C;
  int x = t % Wo;
  int y = (t / Wo) % Ho;
  int nb = t / (Wo * Ho);
  int H = 2 * Ho, W = 2 * Wo;
  const _Float16* base = in + (((size_t)nb * H + 2 * y) * W + 2 * x) * C + ch;
  float a0 = (float)base[0];
  float a1 = (float)base[C];
  float a2 = (float)base[(size_t)W * C];
  float a3 = (float)base[(size_t)W * C + C];
  float m = a0 > a1 ? a0 : a1;
  m = m > a2 ? m : a2;
  m = m > a3 ? m : a3;
  out[idx] = (_Float16)m;
}

// ---------------- weight repack: OIHW f32 -> [Cout][9*Cin] f16 --------------
__global__ void k_wconv_prep(const float* __restrict__ w, _Float16* __restrict__ wT,
                             int Cout, int Cin) {
  int idx = blockIdx.x * blockDim.x + threadIdx.x;
  int K9 = 9 * Cin;
  if (idx >= Cout * K9) return;
  int co = idx / K9;
  int r = idx - co * K9;
  int tap = r / Cin;
  int cin = r - tap * Cin;
  int ky = tap / 3, kx = tap - ky * 3;
  wT[idx] = (_Float16)w[(((size_t)co * Cin + cin) * 3 + ky) * 3 + kx];
}

// f32 [K][N] -> f16 [N][Kp] (zero pad K..Kp)
__global__ void k_kn_to_nkp(const float* __restrict__ in, _Float16* __restrict__ out,
                            int K, int N, int Kp) {
  int idx = blockIdx.x * blockDim.x + threadIdx.x;
  if (idx >= N * Kp) return;
  int n = idx / Kp, k = idx - n * Kp;
  out[idx] = (k < K) ? (_Float16)in[(size_t)k * N + n] : (_Float16)0.f;
}

// f32 [Msrc][K] -> f16 [M][Kp] (zero pad rows>=Msrc and cols>=K)
__global__ void k_mk_to_mkp(const float* __restrict__ in, _Float16* __restrict__ out,
                            int Msrc, int M, int K, int Kp) {
  int idx = blockIdx.x * blockDim.x + threadIdx.x;
  if (idx >= M * Kp) return;
  int m = idx / Kp, k = idx - m * Kp;
  out[idx] = (m < Msrc && k < K) ? (_Float16)in[(size_t)m * K + k] : (_Float16)0.f;
}

// ---------------- graph helpers --------------------------------------------
static __device__ __forceinline__ void edge_sd(const int* __restrict__ ei, int e, int& s, int& d) {
  if (e < NEDGE) { s = ei[e]; d = ei[NEDGE + e]; }
  else { s = d = e - NEDGE; }
}
static __device__ __forceinline__ unsigned fkey(float f) {
  unsigned u = __float_as_uint(f);
  return (u & 0x80000000u) ? ~u : (u | 0x80000000u);
}
static __device__ __forceinline__ float funkey(unsigned k) {
  unsigned u = (k & 0x80000000u) ? (k ^ 0x80000000u) : ~k;
  return __uint_as_float(u);
}

__global__ void k_score(const float* __restrict__ xl, const float* __restrict__ xr,
                        const float* __restrict__ eh, const float* __restrict__ att,
                        const int* __restrict__ ei, float* __restrict__ score) {
  int idx = blockIdx.x * blockDim.x + threadIdx.x;
  if (idx >= NETOT * 3) return;
  int e = idx / 3, h = idx - e * 3;
  int s, d; edge_sd(ei, e, s, d);
  const float* pl = xl + (size_t)s * 768 + h * 256;
  const float* pr = xr + (size_t)d * 768 + h * 256;
  const float* pe = eh + (size_t)e * 768 + h * 256;
  const float* pa = att + h * 256;
  float acc = 0.f;
  for (int c = 0; c < 256; ++c) {
    float m = pl[c] + pr[c] + pe[c];
    m = m > 0.f ? m : 0.2f * m;          // leaky_relu 0.2
    acc += m * pa[c];
  }
  score[idx] = acc;
}

__global__ void k_smax(const float* __restrict__ score, const int* __restrict__ ei,
                       unsigned* __restrict__ smaxu) {
  int idx = blockIdx.x * blockDim.x + threadIdx.x;
  if (idx >= NETOT * 3) return;
  int e = idx / 3, h = idx - e * 3;
  int s, d; edge_sd(ei, e, s, d); (void)s;
  atomicMax(&smaxu[d * 3 + h], fkey(score[idx]));
}

__global__ void k_expden(const float* __restrict__ score, const unsigned* __restrict__ smaxu,
                         const int* __restrict__ ei, float* __restrict__ den,
                         float* __restrict__ extmp) {
  int idx = blockIdx.x * blockDim.x + threadIdx.x;
  if (idx >= NETOT * 3) return;
  int e = idx / 3, h = idx - e * 3;
  int s, d; edge_sd(ei, e, s, d); (void)s;
  float ex = __expf(score[idx] - funkey(smaxu[d * 3 + h]));
  extmp[idx] = ex;
  atomicAdd(&den[d * 3 + h], ex);
}

__global__ void k_alphadiv(float* __restrict__ alpha, const float* __restrict__ den,
                           const int* __restrict__ ei) {
  int idx = blockIdx.x * blockDim.x + threadIdx.x;
  if (idx >= NETOT * 3) return;
  int e = idx / 3, h = idx - e * 3;
  int s, d; edge_sd(ei, e, s, d); (void)s;
  alpha[idx] = alpha[idx] / (den[d * 3 + h] + 1e-16f);
}

__global__ void k_aggregate(const float* __restrict__ alpha, const float* __restrict__ xl,
                            const float* __restrict__ eh, const int* __restrict__ ei,
                            float* __restrict__ outacc) {
  int idx = blockIdx.x * blockDim.x + threadIdx.x;
  if (idx >= NETOT * 768) return;
  int e = idx / 768, r = idx - e * 768;
  int h = r >> 8;
  int s, d; edge_sd(ei, e, s, d);
  float v = alpha[e * 3 + h] * (xl[(size_t)s * 768 + r] + eh[(size_t)e * 768 + r]);
  atomicAdd(&outacc[(size_t)d * 768 + r], v);
}

__global__ void k_mean_heads(const float* __restrict__ acc, float* __restrict__ out,
                             _Float16* __restrict__ outh) {
  int idx = blockIdx.x * blockDim.x + threadIdx.x;
  if (idx >= NBATCH * 256) return;
  int n = idx >> 8, c = idx & 255;
  const float* p = acc + (size_t)n * 768 + c;
  float s = (p[0] + p[256] + p[512]) * (1.f / 3.f);
  out[idx] = s;
  outh[idx] = (_Float16)s;
}

// small VALU GEMM: C[M,N] = A[M,K] (f32) * B[K,N] (f32) — used for vocab head
__global__ void k_small_gemm(const float* __restrict__ A, const float* __restrict__ B,
                             float* __restrict__ C, int M, int K, int N) {
  int idx = blockIdx.x * blockDim.x + threadIdx.x;
  if (idx >= M * N) return;
  int m = idx / N, j = idx - m * N;
  float acc = 0.f;
  for (int k = 0; k < K; ++k) acc += A[(size_t)m * K + k] * B[(size_t)k * N + j];
  C[idx] = acc;
}

__global__ void k_fill_u32(unsigned* __restrict__ p, unsigned v, int n) {
  int idx = blockIdx.x * blockDim.x + threadIdx.x;
  if (idx < n) p[idx] = v;
}
__global__ void k_copy_u32(const unsigned* __restrict__ src, unsigned* __restrict__ dst, int n) {
  int idx = blockIdx.x * blockDim.x + threadIdx.x;
  if (idx < n) dst[idx] = src[idx];
}

// ---------------------------------------------------------------------------
static inline int cdiv(long long a, long long b) { return (int)((a + b - 1) / b); }

extern "C" void kernel_launch(void* const* d_in, const int* in_sizes, int n_in,
                              void* d_out, int out_size, void* d_ws, size_t ws_size,
                              hipStream_t stream) {
  (void)in_sizes; (void)n_in; (void)out_size; (void)ws_size;
  // ---- inputs (setup_inputs insertion order, params flattened recursively) --
  const float* x       = (const float*)d_in[0];
  const int*   ei      = (const int*)d_in[1];
  const float* ea_attr = (const float*)d_in[2];
  const float* cw[8]; const float* cb[8];
  for (int i = 0; i < 8; ++i) { cw[i] = (const float*)d_in[3 + 2 * i]; cb[i] = (const float*)d_in[4 + 2 * i]; }
  const float* fc_w    = (const float*)d_in[19];
  const float* w_vocab = (const float*)d_in[20];
  const float* g_wl[3], *g_wr[3], *g_we[3], *g_att[3], *g_weo[3];
  for (int l = 0; l < 3; ++l) {
    g_wl[l]  = (const float*)d_in[21 + 5 * l + 0];
    g_wr[l]  = (const float*)d_in[21 + 5 * l + 1];
    g_we[l]  = (const float*)d_in[21 + 5 * l + 2];
    g_att[l] = (const float*)d_in[21 + 5 * l + 3];
    g_weo[l] = (const float*)d_in[21 + 5 * l + 4];
  }

  // ---- workspace layout ----------------------------------------------------
  char* ws = (char*)d_ws; size_t o = 0;
  auto take = [&](size_t bytes) { void* p = ws + o; o += (bytes + 255) & ~(size_t)255; return p; };
  _Float16* bufA  = (_Float16*)take(67108864ull * 2);   // ping (134MB), also reused as eh(f32)
  _Float16* bufB  = (_Float16*)take(33554432ull * 2);   // pong (67MB)
  _Float16* wcnv  = (_Float16*)take(2359296ull * 2);    // largest conv weight / fc weight
  _Float16* wg0   = (_Float16*)take(768ull * 256 * 2);  // wl_t
  _Float16* wg1   = (_Float16*)take(768ull * 256 * 2);  // wr_t
  _Float16* wg2   = (_Float16*)take(768ull * 128 * 2);  // we_t
  _Float16* wg3   = (_Float16*)take(128ull * 128 * 2);  // weo_t
  float*    hcur  = (float*)take(262144ull * 4);
  _Float16* hf16  = (_Float16*)take(262144ull * 2);
  float*    eaA   = (float*)take(4325376ull * 4);       // 33792 x 128
  float*    eaB   = (float*)take(4325376ull * 4);
  _Float16* eaf16 = (_Float16*)take(4325376ull * 2);
  float*    xl    = (float*)take(786432ull * 4);
  float*    xr    = (float*)take(786432ull * 4);
  float*    score = (float*)take(101376ull * 4);
  unsigned* smaxu = (unsigned*)take(3072ull * 4);
  float*    den   = (float*)take(3072ull * 4);
  float*    alpha = (float*)take(101376ull * 4);
  float*    outacc= (float*)take(786432ull * 4);
  float*    eh    = (float*)bufA;                        // 33792*768 f32 fits in bufA

  float* out_h     = (float*)d_out;                      // [1024,256]
  int*   out_ei    = (int*)d_out + 262144;               // [2,32768]
  float* out_ea    = (float*)d_out + 327680;             // [32768,128]
  float* out_score = (float*)d_out + 4521984;            // [1024,101]

  // ---- helpers -------------------------------------------------------------
  auto run_gemm = [&](const _Float16* A, const _Float16* Bt, float* C, _Float16* Ch,
                      int M, int N, int Kp) {
    long long tiles = (long long)(M / 32) * (N / 32);
    k_gemm_wmma<<<cdiv(tiles * 32, 128), 128, 0, stream>>>(A, Bt, C, Ch, M, N, Kp);
  };
  auto run_conv = [&](const _Float16* in, _Float16* out2, const float* w, const float* b,
                      int H, int W, int Cin, int Cout) {
    int K9 = 9 * Cin;
    k_wconv_prep<<<cdiv((long long)Cout * K9, 256), 256, 0, stream>>>(w, wcnv, Cout, Cin);
    int M = NBATCH * H * W;
    long long tiles = (long long)(M / 32) * (Cout / 32);
    k_conv_wmma<<<cdiv(tiles * 32, 128), 128, 0, stream>>>(in, wcnv, b, out2, M, H, W, Cin, Cout);
  };
  auto run_pool = [&](const _Float16* in, _Float16* out2, int Ho, int Wo, int C) {
    k_pool<<<cdiv((long long)NBATCH * Ho * Wo * C, 256), 256, 0, stream>>>(in, out2, Ho, Wo, C);
  };

  // ---- VGG11 ---------------------------------------------------------------
  k_conv0<<<cdiv((long long)NBATCH * 32 * 32 * 64, 256), 256, 0, stream>>>(x, cw[0], cb[0], bufA);
  run_pool(bufA, bufB, 16, 16, 64);
  run_conv(bufB, bufA, cw[1], cb[1], 16, 16, 64, 128);
  run_pool(bufA, bufB, 8, 8, 128);
  run_conv(bufB, bufA, cw[2], cb[2], 8, 8, 128, 256);
  run_conv(bufA, bufB, cw[3], cb[3], 8, 8, 256, 256);
  run_pool(bufB, bufA, 4, 4, 256);
  run_conv(bufA, bufB, cw[4], cb[4], 4, 4, 256, 512);
  run_conv(bufB, bufA, cw[5], cb[5], 4, 4, 512, 512);
  run_pool(bufA, bufB, 2, 2, 512);
  run_conv(bufB, bufA, cw[6], cb[6], 2, 2, 512, 512);
  run_conv(bufA, bufB, cw[7], cb[7], 2, 2, 512, 512);
  run_pool(bufB, bufA, 1, 1, 512);                       // bufA = [1024,512] f16

  // FC: h = act @ fc_w  (WMMA), also emit f16 copy of h for the GAT GEMMs
  k_kn_to_nkp<<<cdiv(256 * 512, 256), 256, 0, stream>>>(fc_w, wcnv, 512, 256, 512);
  run_gemm(bufA, wcnv, hcur, hf16, 1024, 256, 512);

  // vocab head (tiny): x_conv_score = h @ w_vocab  [1024,101]
  k_small_gemm<<<cdiv(1024 * 101, 256), 256, 0, stream>>>(hcur, w_vocab, out_score, 1024, 256, 101);

  // ---- 3x GATv2 ------------------------------------------------------------
  float* ea_cur = nullptr;
  for (int l = 0; l < 3; ++l) {
    int ein = (l == 0) ? 10 : 128;
    int Kpe = (l == 0) ? 32 : 128;
    k_kn_to_nkp<<<cdiv(768 * 256, 256), 256, 0, stream>>>(g_wl[l], wg0, 256, 768, 256);
    k_kn_to_nkp<<<cdiv(768 * 256, 256), 256, 0, stream>>>(g_wr[l], wg1, 256, 768, 256);
    k_kn_to_nkp<<<cdiv(768 * Kpe, 256), 256, 0, stream>>>(g_we[l], wg2, ein, 768, Kpe);
    k_kn_to_nkp<<<cdiv(128 * Kpe, 256), 256, 0, stream>>>(g_weo[l], wg3, ein, 128, Kpe);
    if (l == 0)
      k_mk_to_mkp<<<cdiv((long long)NETOT * Kpe, 256), 256, 0, stream>>>(ea_attr, eaf16, NEDGE, NETOT, 10, Kpe);
    else
      k_mk_to_mkp<<<cdiv((long long)NETOT * Kpe, 256), 256, 0, stream>>>(ea_cur, eaf16, NETOT, NETOT, 128, Kpe);

    run_gemm(hf16, wg0, xl, nullptr, 1024, 768, 256);    // xl = h @ wl
    run_gemm(hf16, wg1, xr, nullptr, 1024, 768, 256);    // xr = h @ wr
    run_gemm(eaf16, wg2, eh, nullptr, NETOT, 768, Kpe);  // eh = ea @ we

    k_score<<<cdiv(NETOT * 3, 256), 256, 0, stream>>>(xl, xr, eh, g_att[l], ei, score);
    k_fill_u32<<<cdiv(3072, 256), 256, 0, stream>>>(smaxu, 0u, 3072);
    k_smax<<<cdiv(NETOT * 3, 256), 256, 0, stream>>>(score, ei, smaxu);
    k_fill_u32<<<cdiv(3072, 256), 256, 0, stream>>>((unsigned*)den, 0u, 3072);
    k_expden<<<cdiv(NETOT * 3, 256), 256, 0, stream>>>(score, smaxu, ei, den, alpha);
    k_alphadiv<<<cdiv(NETOT * 3, 256), 256, 0, stream>>>(alpha, den, ei);
    k_fill_u32<<<cdiv(786432, 256), 256, 0, stream>>>((unsigned*)outacc, 0u, 786432);
    k_aggregate<<<cdiv((long long)NETOT * 768, 256), 256, 0, stream>>>(alpha, xl, eh, ei, outacc);

    float* hdst = (l == 2) ? out_h : hcur;
    k_mean_heads<<<cdiv(1024 * 256, 256), 256, 0, stream>>>(outacc, hdst, hf16);

    float* ea_next = (l & 1) ? eaB : eaA;
    run_gemm(eaf16, wg3, ea_next, nullptr, NETOT, 128, Kpe);  // ea' = ea @ w_eo
    ea_cur = ea_next;
  }

  // ---- final copies --------------------------------------------------------
  k_copy_u32<<<cdiv(2 * NEDGE, 256), 256, 0, stream>>>((const unsigned*)ei, (unsigned*)out_ei, 2 * NEDGE);
  k_copy_u32<<<cdiv((long long)NEDGE * 128, 256), 256, 0, stream>>>((const unsigned*)ea_cur,
                                                                    (unsigned*)out_ea, NEDGE * 128);
}
